// CharLSTM_60541859004526
// MI455X (gfx1250) — matrix operand order repaired and checked by reading, not compiled, not run on hardware
//
#include <hip/hip_runtime.h>
#include <hip/hip_bf16.h>

// ---------------- dims ----------------
#define NW    8192      // words
#define LMAX  16
#define WED   506
#define WHD   512
#define XK    512       // WED + CHD
#define G4    2048      // 4*WHD
#define CED   6
#define CHD   6
#define TAGS  64

// ---------------- workspace layout (bytes) ----------------
#define OFF_SYNC   ((size_t)0)                 // count @0, phase @4
#define OFF_HBUF   ((size_t)4096)              // 512 f32 = 2 KB
#define OFF_HCHAR  ((size_t)8192)              // 8192*6*4 = 192 KB
#define OFF_XBF    ((size_t)(1u<<20))          // 8192*512 bf16 = 8 MB
#define OFF_WIHBF  ((size_t)(16u<<20))         // 2048*512 bf16 = 2 MB
#define OFF_OUTWBF ((size_t)(19u<<20))         // 64*512 bf16 = 64 KB
#define OFF_HSBF   ((size_t)(20u<<20))         // 8192*512 bf16 = 8 MB
#define OFF_XP     ((size_t)(30u<<20))         // 8192*2048 f32 = 64 MB

typedef __attribute__((ext_vector_type(16))) __bf16 v16bf;
typedef __attribute__((ext_vector_type(8)))  float  v8f;
typedef unsigned int u32x4 __attribute__((ext_vector_type(4)));
typedef int          i32x4 __attribute__((ext_vector_type(4)));
typedef int          i32x8 __attribute__((ext_vector_type(8)));

union FragBF { v16bf v; uint4 q[2]; };

__device__ __forceinline__ unsigned short f2bf(float f) {
    unsigned int u = __float_as_uint(f);
    unsigned int r = u + 0x7FFFu + ((u >> 16) & 1u);
    return (unsigned short)(r >> 16);
}
__device__ __forceinline__ float sigf(float x) { return 1.0f / (1.0f + __expf(-x)); }

// -------- Tensor Data Mover: 2D tile (bf16) global -> LDS ------------------
// D# per CDNA5 ISA 8.3/8.4: group0 = {count, lds_addr, global_addr, type=2},
// group1 = {data_size, tensor_dim0/1, tile_dim0/1, tensor_dim0_stride}.
// Remaining groups zero (tile_dim3/4 = 0 -> unused). cpol = 0.
// Toolchain exposes the 6-arg builtin:
//   (u32x4 g0, i32x8 g1, i32x4 g2, i32x4 g3, i32x8 gx, i32 cpol)
// Group1 is loop-invariant for a GEMM K-loop, so build it once and reuse.
__device__ __forceinline__ i32x8 tdm_make_g1_bf16(unsigned tile_d0,   // elems
                                                  unsigned tile_d1,   // rows
                                                  unsigned tensor_d0, // elems
                                                  unsigned tensor_d1, // rows
                                                  unsigned stride_d0) // elems
{
    i32x8 g1;
    g1[0] = (int)(1u << 16);                            // data_size=1 (2 bytes)
    g1[1] = (int)((tensor_d0 & 0xFFFFu) << 16);         // tensor_dim0[15:0]
    g1[2] = (int)((tensor_d0 >> 16) |
                  ((tensor_d1 & 0xFFFFu) << 16));       // t_d0[31:16], t_d1[15:0]
    g1[3] = (int)((tensor_d1 >> 16) | (tile_d0 << 16)); // t_d1[31:16], tile_d0
    g1[4] = (int)tile_d1;                               // tile_dim1, tile_dim2=0
    g1[5] = (int)stride_d0;                             // tensor_dim0_stride[31:0]
    g1[6] = 0;                                          // stride hi, dim1_stride lo
    g1[7] = 0;
    return g1;
}

__device__ __forceinline__ void tdm_issue(unsigned lds_addr, const void* gptr,
                                          i32x8 g1) {
    const unsigned long long ga = (unsigned long long)gptr;
    u32x4 g0;
    g0.x = 1u;                                          // count=1 (valid D#)
    g0.y = lds_addr;                                    // lds_addr[63:32]
    g0.z = (unsigned)ga;                                // global_addr[95:64]
    g0.w = ((unsigned)(ga >> 32) & 0x01FFFFFFu)         // global_addr[120:96]
           | (2u << 30);                                // type=2 ("image")
    const i32x4 gz4 = {0, 0, 0, 0};
    const i32x8 gz8 = {0, 0, 0, 0, 0, 0, 0, 0};
    __builtin_amdgcn_tensor_load_to_lds(g0, g1, gz4, gz4, gz8, 0);
}

__device__ __forceinline__ unsigned lds_off(const void* p) {
    return (unsigned)(unsigned long long)(size_t)p;     // LDS aperture: addr[31:0]
}

// ================= kernel 1: char-level LSTM (one thread per word) =========
__global__ void char_lstm_kernel(const int* __restrict__ chars,
                                 const int* __restrict__ lens,
                                 const float* __restrict__ cemb,
                                 const float* __restrict__ Wih,
                                 const float* __restrict__ Whh,
                                 const float* __restrict__ bias,
                                 float* __restrict__ hchar) {
    __shared__ float sWih[4 * CHD * CED];
    __shared__ float sWhh[4 * CHD * CHD];
    __shared__ float sb[4 * CHD];
    const int tid = threadIdx.x;
    for (int i = tid; i < 4 * CHD * CED; i += blockDim.x) sWih[i] = Wih[i];
    for (int i = tid; i < 4 * CHD * CHD; i += blockDim.x) sWhh[i] = Whh[i];
    if (tid < 4 * CHD) sb[tid] = bias[tid];
    __syncthreads();

    const int w = blockIdx.x * blockDim.x + tid;
    float h[CHD] = {0.f, 0.f, 0.f, 0.f, 0.f, 0.f};
    float c[CHD] = {0.f, 0.f, 0.f, 0.f, 0.f, 0.f};
    const int len = lens[w];
    for (int l = 0; l < LMAX; ++l) {
        if (l >= len) break;                       // mask is a prefix: early exit
        const int ch = chars[w * LMAX + l];
        float x[CED];
        #pragma unroll
        for (int e = 0; e < CED; ++e) x[e] = cemb[ch * CED + e];
        float pre[4 * CHD];
        #pragma unroll
        for (int g = 0; g < 4 * CHD; ++g) {
            float p = sb[g];
            #pragma unroll
            for (int e = 0; e < CED; ++e) p += sWih[g * CED + e] * x[e];
            #pragma unroll
            for (int e = 0; e < CHD; ++e) p += sWhh[g * CHD + e] * h[e];
            pre[g] = p;
        }
        #pragma unroll
        for (int j = 0; j < CHD; ++j) {
            float iv = sigf(pre[j]);
            float fv = sigf(pre[CHD + j]);
            float gv = tanhf(pre[2 * CHD + j]);
            float ov = sigf(pre[3 * CHD + j]);
            c[j] = fv * c[j] + iv * gv;
            h[j] = ov * tanhf(c[j]);
        }
    }
    #pragma unroll
    for (int j = 0; j < CHD; ++j) hchar[w * CHD + j] = h[j];
}

// ================= kernel 2: pack X = [word_emb | h_char] -> bf16 ==========
__global__ void pack_x_kernel(const int* __restrict__ words,
                              const float* __restrict__ wemb,
                              const float* __restrict__ hchar,
                              unsigned short* __restrict__ Xbf) {
    const int idx = blockIdx.x * blockDim.x + threadIdx.x;   // NW*XK
    const int row = idx >> 9;
    const int col = idx & (XK - 1);
    float v;
    if (col < WED) v = wemb[(size_t)words[row] * WED + col];
    else           v = hchar[row * CHD + (col - WED)];
    Xbf[idx] = f2bf(v);
}

// ================= kernel 3: f32 -> bf16 converter ========================
__global__ void cvt_bf_kernel(const float* __restrict__ src,
                              unsigned short* __restrict__ dst, int n) {
    const int i = blockIdx.x * blockDim.x + threadIdx.x;
    if (i < n) dst[i] = f2bf(src[i]);
}

// ================= kernel 4: xp = X @ Wih^T + b  (bf16 WMMA + TDM) =========
// M=8192 K=512 N=2048. Block: 256 thr (8 waves), 128x128 tile, wave 32x64.
// Tiles staged by the Tensor Data Mover: wave0 -> A, wave1 -> B.
__global__ void __launch_bounds__(256, 1)
gemm_xp_kernel(const unsigned short* __restrict__ Xbf,
               const unsigned short* __restrict__ Wbf,
               const float* __restrict__ wb,
               float* __restrict__ xp) {
    __shared__ unsigned short As[128 * 32];
    __shared__ unsigned short Bs[128 * 32];
    const int tid = threadIdx.x;
    const int lane = tid & 31;
    const int wv = tid >> 5;
    const int warpM = wv & 3;          // 4 waves along M
    const int warpN = wv >> 2;         // 2 waves along N
    const int bm = blockIdx.x * 128;
    const int bn = blockIdx.y * 128;

    v8f acc[2][4];
    const v8f vz = {0.f, 0.f, 0.f, 0.f, 0.f, 0.f, 0.f, 0.f};
    #pragma unroll
    for (int mi = 0; mi < 2; ++mi)
        #pragma unroll
        for (int ni = 0; ni < 4; ++ni) acc[mi][ni] = vz;

    const unsigned as_addr = lds_off(As);
    const unsigned bs_addr = lds_off(Bs);
    const int mr = lane & 15;
    const int hk = lane >> 4;

    // loop-invariant descriptor group1 (tile 32x128, row stride XK)
    const i32x8 g1a = tdm_make_g1_bf16(32, 128, XK, NW, XK);
    const i32x8 g1b = tdm_make_g1_bf16(32, 128, XK, G4, XK);

    // prologue DMA: first K tile
    if (wv == 0)      tdm_issue(as_addr, Xbf + (size_t)bm * XK, g1a);
    else if (wv == 1) tdm_issue(bs_addr, Wbf + (size_t)bn * XK, g1b);

    for (int k0 = 0; k0 < XK; k0 += 32) {
        __builtin_amdgcn_s_wait_tensorcnt(0);
        __syncthreads();                       // LDS tile visible to all waves

        FragBF af[2], bf[4];
        #pragma unroll
        for (int mi = 0; mi < 2; ++mi) {
            const unsigned short* ap = As + (warpM * 32 + mi * 16 + mr) * 32;
            af[mi].q[0] = *(const uint4*)(ap + hk * 8);
            af[mi].q[1] = *(const uint4*)(ap + hk * 8 + 16);
        }
        #pragma unroll
        for (int ni = 0; ni < 4; ++ni) {
            const uint4* bp = (const uint4*)(Bs + (warpN * 64 + ni * 16 + mr) * 32 + hk * 8);
            bf[ni].q[0] = bp[0];
            bf[ni].q[1] = bp[2];
        }
        asm volatile("s_wait_dscnt 0x0" ::: "memory");  // frags landed in VGPRs
        __syncthreads();                       // tile fully consumed

        if (k0 + 32 < XK) {                    // overlap next DMA with WMMAs
            if (wv == 0)
                tdm_issue(as_addr, Xbf + (size_t)bm * XK + k0 + 32, g1a);
            else if (wv == 1)
                tdm_issue(bs_addr, Wbf + (size_t)bn * XK + k0 + 32, g1b);
        }

        #pragma unroll
        for (int mi = 0; mi < 2; ++mi)
            #pragma unroll
            for (int ni = 0; ni < 4; ++ni)
                acc[mi][ni] = __builtin_amdgcn_wmma_f32_16x16x32_bf16(
                    false, af[mi].v, false, bf[ni].v, (short)0, acc[mi][ni], false, false);
    }

    // epilogue: add bias, store fp32 xp
    #pragma unroll
    for (int mi = 0; mi < 2; ++mi) {
        #pragma unroll
        for (int ni = 0; ni < 4; ++ni) {
            const int row = bm + warpM * 32 + mi * 16 + hk * 8;
            const int col = bn + warpN * 64 + ni * 16 + mr;
            const float b = wb[col];
            #pragma unroll
            for (int r = 0; r < 8; ++r)
                xp[(size_t)(row + r) * G4 + col] = acc[mi][ni][r] + b;
        }
    }
}

// ================= kernel 5: persistent word-LSTM recurrence ===============
// 16 blocks x 128 threads. Block g owns hidden units j in [32g, 32g+32) and
// Whh rows {j, 512+j, 1024+j, 1536+j} resident in LDS (padded stride 516).
#define WROWS 128
#define WSTRIDE 516
#define NBLK 16
__global__ void __launch_bounds__(128, 1)
word_lstm_kernel(const float* __restrict__ Whh,
                 const float* __restrict__ xp,
                 float* __restrict__ hbuf,
                 unsigned short* __restrict__ hs_bf,
                 int* __restrict__ cnt,
                 int* __restrict__ phs) {
    extern __shared__ float smem[];
    float* sW   = smem;                         // 128 * 516
    float* h_s  = smem + WROWS * WSTRIDE;       // 512
    float* gbuf = h_s + WHD;                    // 128

    const int tid  = threadIdx.x;
    const int gate = tid >> 5;
    const int jj   = tid & 31;
    const int jglob = (blockIdx.x << 5) + jj;
    const int grow  = gate * WHD + jglob;       // row of Whh / index into pre

    // one-time: load my Whh row into LDS
    {
        const float4* src = (const float4*)(Whh + (size_t)grow * WHD);
        float4* dst = (float4*)(sW + tid * WSTRIDE);
        #pragma unroll 4
        for (int i = 0; i < WHD / 4; ++i) dst[i] = src[i];
    }
    __syncthreads();

    float cst = 0.f;                            // cell state (valid in tid<32)
    for (int t = 0; t < NW; ++t) {
        // pull h (device-coherent) into LDS
        for (int k = tid; k < WHD; k += WROWS)
            h_s[k] = __hip_atomic_load(&hbuf[k], __ATOMIC_RELAXED, __HIP_MEMORY_SCOPE_AGENT);
        __syncthreads();

        __builtin_prefetch(xp + (size_t)(t + 1) * G4 + grow, 0, 1);

        const float* wr = sW + tid * WSTRIDE;
        float acc = xp[(size_t)t * G4 + grow];
        #pragma unroll 8
        for (int k = 0; k < WHD; ++k) acc += wr[k] * h_s[k];

        gbuf[tid] = (gate == 2) ? tanhf(acc) : sigf(acc);
        __syncthreads();

        if (tid < 32) {
            const float iv = gbuf[tid];
            const float fv = gbuf[32 + tid];
            const float gv = gbuf[64 + tid];
            const float ov = gbuf[96 + tid];
            cst = fv * cst + iv * gv;
            const float hv = ov * tanhf(cst);
            __hip_atomic_store(&hbuf[jglob], hv, __ATOMIC_RELAXED, __HIP_MEMORY_SCOPE_AGENT);
            hs_bf[(size_t)t * WHD + jglob] = f2bf(hv);
        }
        __syncthreads();

        // grid-wide barrier (monotonic phase)
        if (tid == 0) {
            __threadfence();
            const int prev = atomicAdd(cnt, 1);
            if (prev == NBLK - 1) {
                atomicExch(cnt, 0);
                __threadfence();
                atomicAdd(phs, 1);
            } else {
                while (__hip_atomic_load(phs, __ATOMIC_RELAXED, __HIP_MEMORY_SCOPE_AGENT) < t + 1)
                    __builtin_amdgcn_s_sleep(2);
            }
        }
        __syncthreads();
    }
}

// ================= kernel 6: logits GEMM (WMMA + TDM) + log_softmax ========
// hs[8192,512] bf16 @ outW[64,512]^T bf16 -> [8192,64]; per-row log_softmax.
__global__ void __launch_bounds__(256, 1)
logits_kernel(const unsigned short* __restrict__ hs,
              const unsigned short* __restrict__ Wo,
              const float* __restrict__ ob,
              float* __restrict__ out) {
    __shared__ unsigned short As[128 * 32];
    __shared__ unsigned short Bs[64 * 32];
    __shared__ float Cs[128 * 64];
    const int tid = threadIdx.x;
    const int lane = tid & 31;
    const int wv = tid >> 5;           // 8 waves, each 16 rows x 64 cols
    const int bm = blockIdx.x * 128;
    const int mr = lane & 15;
    const int hk = lane >> 4;

    v8f acc[4];
    const v8f vz = {0.f, 0.f, 0.f, 0.f, 0.f, 0.f, 0.f, 0.f};
    #pragma unroll
    for (int ni = 0; ni < 4; ++ni) acc[ni] = vz;

    const unsigned as_addr = lds_off(As);
    const unsigned bs_addr = lds_off(Bs);

    const i32x8 g1a = tdm_make_g1_bf16(32, 128, WHD, NW, WHD);
    const i32x8 g1b = tdm_make_g1_bf16(32, 64, WHD, TAGS, WHD);

    if (wv == 0)      tdm_issue(as_addr, hs + (size_t)bm * WHD, g1a);
    else if (wv == 1) tdm_issue(bs_addr, Wo, g1b);

    for (int k0 = 0; k0 < WHD; k0 += 32) {
        __builtin_amdgcn_s_wait_tensorcnt(0);
        __syncthreads();

        FragBF a, b[4];
        const unsigned short* ap = As + (wv * 16 + mr) * 32;
        a.q[0] = *(const uint4*)(ap + hk * 8);
        a.q[1] = *(const uint4*)(ap + hk * 8 + 16);
        #pragma unroll
        for (int ni = 0; ni < 4; ++ni) {
            const uint4* bp = (const uint4*)(Bs + (ni * 16 + mr) * 32 + hk * 8);
            b[ni].q[0] = bp[0];
            b[ni].q[1] = bp[2];
        }
        asm volatile("s_wait_dscnt 0x0" ::: "memory");
        __syncthreads();

        if (k0 + 32 < WHD) {
            if (wv == 0)
                tdm_issue(as_addr, hs + (size_t)bm * WHD + k0 + 32, g1a);
            else if (wv == 1)
                tdm_issue(bs_addr, Wo + k0 + 32, g1b);
        }

        #pragma unroll
        for (int ni = 0; ni < 4; ++ni)
            acc[ni] = __builtin_amdgcn_wmma_f32_16x16x32_bf16(
                false, a.v, false, b[ni].v, (short)0, acc[ni], false, false);
    }

    #pragma unroll
    for (int ni = 0; ni < 4; ++ni) {
        const float b = ob[ni * 16 + mr];
        #pragma unroll
        for (int r = 0; r < 8; ++r)
            Cs[(wv * 16 + hk * 8 + r) * TAGS + ni * 16 + mr] = acc[ni][r] + b;
    }
    __syncthreads();

    if (tid < 128) {
        const float* rowp = Cs + tid * TAGS;
        float m = -1e30f;
        for (int c = 0; c < TAGS; ++c) m = fmaxf(m, rowp[c]);
        float s = 0.f;
        for (int c = 0; c < TAGS; ++c) s += __expf(rowp[c] - m);
        const float l = __logf(s);
        float* o = out + (size_t)(bm + tid) * TAGS;
        for (int c = 0; c < TAGS; ++c) o[c] = rowp[c] - m - l;
    }
}

// ================= host launcher ==========================================
extern "C" void kernel_launch(void* const* d_in, const int* in_sizes, int n_in,
                              void* d_out, int out_size, void* d_ws, size_t ws_size,
                              hipStream_t stream) {
    const int*   words = (const int*)d_in[0];
    const int*   chars = (const int*)d_in[1];
    const int*   lens  = (const int*)d_in[2];
    const float* wemb  = (const float*)d_in[3];
    const float* cemb  = (const float*)d_in[4];
    const float* cWih  = (const float*)d_in[5];
    const float* cWhh  = (const float*)d_in[6];
    const float* cb    = (const float*)d_in[7];
    const float* wWih  = (const float*)d_in[8];
    const float* wWhh  = (const float*)d_in[9];
    const float* wb    = (const float*)d_in[10];
    const float* outW  = (const float*)d_in[11];
    const float* outb  = (const float*)d_in[12];
    float* out = (float*)d_out;

    char* ws = (char*)d_ws;
    int*            sync_cnt = (int*)(ws + OFF_SYNC);
    int*            sync_phs = (int*)(ws + OFF_SYNC) + 1;
    float*          hbuf     = (float*)(ws + OFF_HBUF);
    float*          hchar    = (float*)(ws + OFF_HCHAR);
    unsigned short* Xbf      = (unsigned short*)(ws + OFF_XBF);
    unsigned short* WihBf    = (unsigned short*)(ws + OFF_WIHBF);
    unsigned short* outWBf   = (unsigned short*)(ws + OFF_OUTWBF);
    unsigned short* hsBf     = (unsigned short*)(ws + OFF_HSBF);
    float*          xp       = (float*)(ws + OFF_XP);

    // zero barrier state + initial h (covers [0, 6144))
    (void)hipMemsetAsync(ws, 0, OFF_HBUF + WHD * sizeof(float), stream);

    // 1) char-level LSTM
    char_lstm_kernel<<<NW / 128, 128, 0, stream>>>(chars, lens, cemb, cWih, cWhh, cb, hchar);

    // 2) pack X (bf16) and convert weights to bf16
    pack_x_kernel<<<(NW * XK) / 256, 256, 0, stream>>>(words, wemb, hchar, Xbf);
    cvt_bf_kernel<<<(G4 * XK) / 256, 256, 0, stream>>>(wWih, WihBf, G4 * XK);
    cvt_bf_kernel<<<(TAGS * WHD) / 256, 256, 0, stream>>>(outW, outWBf, TAGS * WHD);

    // 3) big input-projection GEMM via WMMA + TDM
    gemm_xp_kernel<<<dim3(NW / 128, G4 / 128), 256, 0, stream>>>(Xbf, WihBf, wb, xp);

    // 4) persistent word-LSTM recurrence (16 co-resident WGs, Whh in LDS)
    const int smem_word = (WROWS * WSTRIDE + WHD + WROWS) * (int)sizeof(float);
    (void)hipFuncSetAttribute((const void*)word_lstm_kernel,
                              hipFuncAttributeMaxDynamicSharedMemorySize, smem_word);
    word_lstm_kernel<<<NBLK, WROWS, smem_word, stream>>>(wWhh, xp, hbuf, hsBf,
                                                         sync_cnt, sync_phs);

    // 5) logits GEMM (WMMA + TDM) + log_softmax
    logits_kernel<<<NW / 128, 256, 0, stream>>>(hsBf, outWBf, outb, out);
}
